// CrossAttention_34866544509741
// MI455X (gfx1250) — compile-verified, matrix operand order checked by reading, not run on hardware
//
#include <hip/hip_runtime.h>

// ---------------- constants ----------------
#define DIMC   192
#define HEADS  6
#define HD     32
#define TOK    64
#define NWIN   64
#define TT     (TOK * DIMC)     // 12288 elements per window for x/k/v/out
#define WSZ    (DIMC * DIMC)    // 36864 weight elements
#define SCALE  0.17677669529663687f  // 32^-0.5

typedef _Float16 half4_t __attribute__((ext_vector_type(4)));
typedef _Float16 half8_t __attribute__((ext_vector_type(8)));
typedef _Float16 v16h    __attribute__((ext_vector_type(16)));
typedef float    v8f     __attribute__((ext_vector_type(8)));

union FragU { v16h v; half8_t h[2]; };

// LDS offsets in halfs
#define OFF_X  0        // X (64x192) f16; later reused as O
#define OFF_Q  12288    // Q (64x192) f16, pre-scaled, head h in cols [32h,32h+32)
#define OFF_K  24576    // K [h][t][d] f16
#define OFF_V  36864    // V transposed [h][d][t] f16
#define OFF_P  49152    // P per-wave [6][64][64] f16
#define OFF_W  73728    // W (192x192) f16 (q_w, then proj_w)
#define LDS_HALFS 110592
#define LDS_BYTES (LDS_HALFS * 2)   // 221184 B <= 320KB WGP LDS

// Load a 16x(32) f16 fragment per ISA 7.12.2:
// lane l holds row (row0 + l%16); halfs j: K = (j&7) + 8*g + 2*(j&8), g = l/16.
// => two contiguous 8-half (16B) chunks at k0+8g and k0+16+8g.
// Used for A (src row-major MxK) and B (src stored transposed: NxK row-major).
__device__ __forceinline__ v16h load_frag(const _Float16* p, int lda, int row0,
                                          int k0, int lane) {
  const int r = row0 + (lane & 15);
  const int g = (lane >> 4) & 1;
  const _Float16* base = p + r * lda + k0 + 8 * g;
  FragU f;
  f.h[0] = *(const half8_t*)(base);
  f.h[1] = *(const half8_t*)(base + 16);
  return f.v;
}

__device__ __forceinline__ v8f wmma_f16(v16h a, v16h b, v8f c) {
  return __builtin_amdgcn_wmma_f32_16x16x32_f16(false, a, false, b, (short)0, c,
                                                false, false);
}

__device__ __forceinline__ v8f zero8() {
  v8f z;
#pragma unroll
  for (int r = 0; r < 8; ++r) z[r] = 0.0f;
  return z;
}

__global__ void __launch_bounds__(256, 1)
swin_window_attn(const float* __restrict__ x, const float* __restrict__ k,
                 const float* __restrict__ v, const float* __restrict__ mask,
                 const float* __restrict__ rpb, const float* __restrict__ qw,
                 const float* __restrict__ qb, const float* __restrict__ pw,
                 const float* __restrict__ pb, const int* __restrict__ rel,
                 float* __restrict__ out) {
  extern __shared__ char smem_raw[];
  _Float16* S16 = (_Float16*)smem_raw;
  _Float16* Xh  = S16 + OFF_X;   // also O buffer later
  _Float16* Qh  = S16 + OFF_Q;
  _Float16* Kh  = S16 + OFF_K;
  _Float16* Vth = S16 + OFF_V;
  _Float16* Ph  = S16 + OFF_P;
  _Float16* Wh  = S16 + OFF_W;

  const int b    = blockIdx.x;
  const int tid  = threadIdx.x;
  const int lane = tid & 31;
  const int wv   = tid >> 5;     // 8 waves

  // ---------- Stage 0: cooperative staging into LDS (f32 -> f16) ----------
  {
    const float* xb = x + (size_t)b * TT;
    const float* kb = k + (size_t)b * TT;
    const float* vb = v + (size_t)b * TT;
#pragma unroll
    for (int i4 = tid; i4 < TT / 4; i4 += 256) {
      float4 xv = ((const float4*)xb)[i4];
      half4_t hx = {(_Float16)xv.x, (_Float16)xv.y, (_Float16)xv.z, (_Float16)xv.w};
      *(half4_t*)(Xh + 4 * i4) = hx;
      float4 kv = ((const float4*)kb)[i4];
      half4_t hk = {(_Float16)kv.x, (_Float16)kv.y, (_Float16)kv.z, (_Float16)kv.w};
      *(half4_t*)(Kh + 4 * i4) = hk;
    }
    // V transposed per head: Vth[h][d][t]
    for (int i = tid; i < TT; i += 256) {
      int h = i >> 11;            // / 2048
      int rem = i & 2047;
      int t = rem >> 5;
      int d = rem & 31;
      Vth[h * 2048 + d * TOK + t] = (_Float16)vb[i];
    }
    // q_w (row-major [n_out][k_in], exactly the transposed-B layout we need)
    for (int i4 = tid; i4 < WSZ / 4; i4 += 256) {
      float4 wvv = ((const float4*)qw)[i4];
      half4_t hw = {(_Float16)wvv.x, (_Float16)wvv.y, (_Float16)wvv.z, (_Float16)wvv.w};
      *(half4_t*)(Wh + 4 * i4) = hw;
    }
  }
  __syncthreads();

  // ---------- Stage 1: Q = (X @ qw^T + qb) * SCALE  (48 tiles / 8 waves) ----------
#pragma unroll
  for (int i = 0; i < 6; ++i) {
    const int tt = wv * 6 + i;
    const int mt = tt / 12, nt = tt % 12;
    v8f acc = zero8();
#pragma unroll
    for (int kk = 0; kk < 6; ++kk) {
      v16h a  = load_frag(Xh, DIMC, mt * 16, kk * 32, lane);
      v16h bb = load_frag(Wh, DIMC, nt * 16, kk * 32, lane);
      acc = wmma_f16(a, bb, acc);
    }
    const int m0 = mt * 16 + ((lane >> 4) << 3);
    const int n  = nt * 16 + (lane & 15);
    const float bias = qb[n];
#pragma unroll
    for (int r = 0; r < 8; ++r)
      Qh[(m0 + r) * DIMC + n] = (_Float16)((acc[r] + bias) * SCALE);
  }
  __syncthreads();  // Q ready; q_w no longer needed -> Wh free

  // ---------- Stage 2: per-head attention (waves 0..5); waves 6..7 prefetch proj_w ----------
  if (wv < HEADS) {
    const int h = wv;
    const int w = b & (NWIN - 1);
    const float* mrow = mask + w * (TOK * TOK);
    _Float16* Pw = Ph + h * (TOK * TOK);

    // S accumulators seeded with relative-position bias + shift mask
    v8f S[4][4];
#pragma unroll
    for (int mt = 0; mt < 4; ++mt) {
      const int m0 = mt * 16 + ((lane >> 4) << 3);
#pragma unroll
      for (int nt = 0; nt < 4; ++nt) {
        const int n = nt * 16 + (lane & 15);
#pragma unroll
        for (int r = 0; r < 8; ++r) {
          const int m = m0 + r;
          const int idx = rel[m * TOK + n];
          S[mt][nt][r] = rpb[idx * HEADS + h] + mrow[m * TOK + n];
        }
      }
    }
    // S += Q_h @ K_h^T   (K = head_dim = 32 -> one WMMA per tile)
#pragma unroll
    for (int mt = 0; mt < 4; ++mt) {
      v16h a = load_frag(Qh, DIMC, mt * 16, h * HD, lane);
#pragma unroll
      for (int nt = 0; nt < 4; ++nt) {
        v16h bb = load_frag(Kh + h * (TOK * HD), HD, nt * 16, 0, lane);
        S[mt][nt] = wmma_f16(a, bb, S[mt][nt]);
      }
    }
    // softmax over rows: each row lives in one 16-lane half across 4 nt tiles
#pragma unroll
    for (int mt = 0; mt < 4; ++mt) {
      const int m0 = mt * 16 + ((lane >> 4) << 3);
#pragma unroll
      for (int r = 0; r < 8; ++r) {
        float mx = S[mt][0][r];
#pragma unroll
        for (int nt = 1; nt < 4; ++nt) mx = fmaxf(mx, S[mt][nt][r]);
#pragma unroll
        for (int off = 8; off >= 1; off >>= 1)
          mx = fmaxf(mx, __shfl_xor(mx, off, 32));
        float sm = 0.0f;
#pragma unroll
        for (int nt = 0; nt < 4; ++nt) {
          float e = __expf(S[mt][nt][r] - mx);
          S[mt][nt][r] = e;
          sm += e;
        }
#pragma unroll
        for (int off = 8; off >= 1; off >>= 1)
          sm += __shfl_xor(sm, off, 32);
        const float inv = 1.0f / sm;
        const int m = m0 + r;
#pragma unroll
        for (int nt = 0; nt < 4; ++nt) {
          const int n = nt * 16 + (lane & 15);
          Pw[m * TOK + n] = (_Float16)(S[mt][nt][r] * inv);
        }
      }
    }
    // O_h = P @ V_h  (M=64, N=32, K=64)
    v8f O[4][2];
#pragma unroll
    for (int mt = 0; mt < 4; ++mt)
#pragma unroll
      for (int nv = 0; nv < 2; ++nv) O[mt][nv] = zero8();
#pragma unroll
    for (int mt = 0; mt < 4; ++mt) {
#pragma unroll
      for (int ks = 0; ks < 2; ++ks) {
        v16h a = load_frag(Pw, TOK, mt * 16, ks * 32, lane);
#pragma unroll
        for (int nv = 0; nv < 2; ++nv) {
          v16h bb = load_frag(Vth + h * (HD * TOK), TOK, nv * 16, ks * 32, lane);
          O[mt][nv] = wmma_f16(a, bb, O[mt][nv]);
        }
      }
    }
    // store O_h (f16) into X buffer at columns [32h, 32h+32)
#pragma unroll
    for (int mt = 0; mt < 4; ++mt) {
      const int m0 = mt * 16 + ((lane >> 4) << 3);
#pragma unroll
      for (int nv = 0; nv < 2; ++nv) {
        const int d = nv * 16 + (lane & 15);
#pragma unroll
        for (int r = 0; r < 8; ++r)
          Xh[(m0 + r) * DIMC + h * HD + d] = (_Float16)O[mt][nv][r];
      }
    }
  } else {
    // waves 6,7: overlap proj_w staging with attention
    const int t2 = tid - 192;
    for (int i4 = t2; i4 < WSZ / 4; i4 += 64) {
      float4 wvv = ((const float4*)pw)[i4];
      half4_t hw = {(_Float16)wvv.x, (_Float16)wvv.y, (_Float16)wvv.z, (_Float16)wvv.w};
      *(half4_t*)(Wh + 4 * i4) = hw;
    }
  }
  __syncthreads();

  // ---------- Stage 3: out = O @ proj_w^T + proj_b ----------
  float* ob = out + (size_t)b * TT;
#pragma unroll
  for (int i = 0; i < 6; ++i) {
    const int tt = wv * 6 + i;
    const int mt = tt / 12, nt = tt % 12;
    v8f acc = zero8();
#pragma unroll
    for (int kk = 0; kk < 6; ++kk) {
      v16h a  = load_frag(Xh, DIMC, mt * 16, kk * 32, lane);
      v16h bb = load_frag(Wh, DIMC, nt * 16, kk * 32, lane);
      acc = wmma_f16(a, bb, acc);
    }
    const int m0 = mt * 16 + ((lane >> 4) << 3);
    const int n  = nt * 16 + (lane & 15);
    const float bias = pb[n];
#pragma unroll
    for (int r = 0; r < 8; ++r)
      ob[(m0 + r) * DIMC + n] = acc[r] + bias;
  }
}

extern "C" void kernel_launch(void* const* d_in, const int* in_sizes, int n_in,
                              void* d_out, int out_size, void* d_ws, size_t ws_size,
                              hipStream_t stream) {
  const float* x    = (const float*)d_in[0];
  const float* k    = (const float*)d_in[1];
  const float* v    = (const float*)d_in[2];
  const float* mask = (const float*)d_in[3];
  const float* rpb  = (const float*)d_in[4];
  const float* qw   = (const float*)d_in[5];
  const float* qb   = (const float*)d_in[6];
  const float* pw   = (const float*)d_in[7];
  const float* pb   = (const float*)d_in[8];
  const int*   rel  = (const int*)d_in[9];
  float* out = (float*)d_out;

  const int bw = in_sizes[0] / TT;  // 8192

  hipFuncSetAttribute((const void*)swin_window_attn,
                      hipFuncAttributeMaxDynamicSharedMemorySize, LDS_BYTES);
  swin_window_attn<<<bw, 256, LDS_BYTES, stream>>>(x, k, v, mask, rpb, qw, qb,
                                                   pw, pb, rel, out);
}
